// PPR_25744033972455
// MI455X (gfx1250) — matrix operand order repaired and checked by reading, not compiled, not run on hardware
//
#include <hip/hip_runtime.h>
#include <hip/hip_bf16.h>

// ---------------------------------------------------------------------------
// PPR power series on MI455X (gfx1250).
//   out = th0*P0 + th1*A + th2*thr(A@A) + th3*thr(A@thr(A@A))
// Dense 4096x4096, two 4096^3 GEMMs (275 GFLOP) -> compute bound on the
// matrix pipe; whole working set (<=160MB) is L2-resident (192MB L2).
// GEMMs run on V_WMMA_F32_16X16X32_F16 (f16 in, f32 accumulate).
// ---------------------------------------------------------------------------

typedef __attribute__((ext_vector_type(16))) _Float16 v16h;
typedef __attribute__((ext_vector_type(8)))  _Float16 v8h;
typedef __attribute__((ext_vector_type(8)))  float    v8f;

constexpr int   NN   = 4096;
constexpr float EPS  = 0.005f;
constexpr float TH0  = 0.4f;                 // alpha
constexpr float TH1  = 0.4f * 0.6f;          // 0.24
constexpr float TH2  = 0.4f * 0.6f * 0.6f;   // 0.144
constexpr float TH3  = TH2 * 0.6f;           // 0.0864

// ---------------------------------------------------------------- zero init
__global__ void ppr_zero2(float* __restrict__ a, float* __restrict__ b) {
    int i = blockIdx.x * blockDim.x + threadIdx.x;   // grid sized exactly
    float4 z = {0.f, 0.f, 0.f, 0.f};
    reinterpret_cast<float4*>(a)[i] = z;
    reinterpret_cast<float4*>(b)[i] = z;
}

// ------------------------------------------------------------- edge scatter
// A[src,dst] += w ; out[src,dst] += th0*1 + th1*w   (duplicates accumulate,
// matching .at[].add semantics for both P0 and A).
__global__ void ppr_scatter(const long long* __restrict__ ei,
                            const float* __restrict__ attr,
                            float* __restrict__ Af32,
                            float* __restrict__ out, int E) {
    int e = blockIdx.x * blockDim.x + threadIdx.x;
    if (e >= E) return;
    int s = (int)ei[e];
    int d = (int)ei[(size_t)E + e];
    float w = attr[e];
    size_t idx = (size_t)s * NN + d;
    atomicAdd(Af32 + idx, w);
    atomicAdd(out  + idx, TH0 + TH1 * w);
}

// --------------------------------------------- f32 -> f16 convert + transpose
// Produces Ah (row major, WMMA A operand) and AhT (transpose, WMMA B operand:
// each lane reads two runs of 8 contiguous K at fixed column). 64x64 LDS tile, all
// global traffic coalesced.
__global__ void ppr_convert(const float* __restrict__ Af32,
                            _Float16* __restrict__ Ah,
                            _Float16* __restrict__ AhT) {
    __shared__ float tile[64][65];
    int tm = blockIdx.y, tn = blockIdx.x;
#pragma unroll
    for (int i = 0; i < 16; ++i) {
        int idx = threadIdx.x + 256 * i;
        int r = idx >> 6, c = idx & 63;
        float v = Af32[(size_t)(tm * 64 + r) * NN + tn * 64 + c];
        tile[r][c] = v;
        Ah[(size_t)(tm * 64 + r) * NN + tn * 64 + c] = (_Float16)v;
    }
    __syncthreads();
#pragma unroll
    for (int i = 0; i < 16; ++i) {
        int idx = threadIdx.x + 256 * i;
        int r = idx >> 6, c = idx & 63;
        AhT[(size_t)(tn * 64 + r) * NN + tm * 64 + c] = (_Float16)tile[c][r];
    }
}

// ------------------------------------------------------------------- GEMM
// C = Ah @ Bt^T  (Bt holds B transposed, row-major [n][k]).
// Block = 256 threads = 8 waves, block tile 128x128, wave tile 64x32
// (4x2 grid of 16x16 WMMA fragments, 8 v_wmma per 32-wide K step).
// Epilogue: t = thr(c);  out[m][n] += theta*t;  optionally M2T[n][m] = (f16)t
// (C-fragment rows are contiguous per lane -> single b128 store of the
//  transposed f16 tile, which is exactly GEMM2's B-operand layout).
template <bool WRITE_M2T>
__global__ __launch_bounds__(256)
void ppr_gemm(const _Float16* __restrict__ Ah,
              const _Float16* __restrict__ Bt,
              _Float16* __restrict__ M2T,
              float* __restrict__ out, float theta) {
    const int lane  = threadIdx.x & 31;
    const int wave  = threadIdx.x >> 5;
    const int wm    = wave >> 2;            // 0..1
    const int wn    = wave & 3;             // 0..3
    const int m0    = blockIdx.y * 128 + wm * 64;
    const int n0    = blockIdx.x * 128 + wn * 32;
    const int mlane = lane & 15;
    const int half  = lane >> 4;

    v8f acc[4][2];
#pragma unroll
    for (int fm = 0; fm < 4; ++fm)
#pragma unroll
        for (int fn = 0; fn < 2; ++fn) acc[fm][fn] = (v8f){};

    const _Float16* aptr[4];
    const _Float16* bptr[2];
#pragma unroll
    for (int fm = 0; fm < 4; ++fm)
        aptr[fm] = Ah + (size_t)(m0 + fm * 16 + mlane) * NN;
#pragma unroll
    for (int fn = 0; fn < 2; ++fn)
        bptr[fn] = Bt + (size_t)(n0 + fn * 16 + mlane) * NN;

    for (int k = 0; k < NN; k += 32) {
        v16h a[4], b[2];
#pragma unroll
        for (int fm = 0; fm < 4; ++fm) {
            // A 16x32 f16 fragment: lane holds K = 8*half+0..7, 16+8*half+0..7
            v8h lo = *reinterpret_cast<const v8h*>(aptr[fm] + k + half * 8);
            v8h hi = *reinterpret_cast<const v8h*>(aptr[fm] + k + 16 + half * 8);
            a[fm] = __builtin_shufflevector(lo, hi, 0, 1, 2, 3, 4, 5, 6, 7,
                                            8, 9, 10, 11, 12, 13, 14, 15);
        }
#pragma unroll
        for (int fn = 0; fn < 2; ++fn) {
            // B 32x16 f16 fragment: lane holds K = 8*half+0..7, 16+8*half+0..7
            v8h lo = *reinterpret_cast<const v8h*>(bptr[fn] + k + half * 8);
            v8h hi = *reinterpret_cast<const v8h*>(bptr[fn] + k + 16 + half * 8);
            b[fn] = __builtin_shufflevector(lo, hi, 0, 1, 2, 3, 4, 5, 6, 7,
                                            8, 9, 10, 11, 12, 13, 14, 15);
        }
#pragma unroll
        for (int fm = 0; fm < 4; ++fm)
#pragma unroll
            for (int fn = 0; fn < 2; ++fn)
                acc[fm][fn] = __builtin_amdgcn_wmma_f32_16x16x32_f16(
                    false, a[fm], false, b[fn], (short)0, acc[fm][fn],
                    false, false);
    }

    // ------------------------------------------------------------ epilogue
#pragma unroll
    for (int fm = 0; fm < 4; ++fm) {
#pragma unroll
        for (int fn = 0; fn < 2; ++fn) {
            v8f c = acc[fm][fn];
            int n     = n0 + fn * 16 + mlane;
            int mbase = m0 + fm * 16 + half * 8;   // 8 consecutive rows/lane
            if (WRITE_M2T) {
                v8h pack;
#pragma unroll
                for (int v = 0; v < 8; ++v) {
                    float t = (c[v] >= EPS) ? c[v] : 0.f;
                    pack[v] = (_Float16)t;
                    out[(size_t)(mbase + v) * NN + n] += theta * t;
                }
                *reinterpret_cast<v8h*>(M2T + (size_t)n * NN + mbase) = pack;
            } else {
#pragma unroll
                for (int v = 0; v < 8; ++v) {
                    float t = (c[v] >= EPS) ? c[v] : 0.f;
                    out[(size_t)(mbase + v) * NN + n] += theta * t;
                }
            }
        }
    }
}

// ---------------------------------------------------------------------------
extern "C" void kernel_launch(void* const* d_in, const int* in_sizes, int n_in,
                              void* d_out, int out_size, void* d_ws, size_t ws_size,
                              hipStream_t stream) {
    // inputs: x (unused), edge_index int64 [2,E], edge_attr f32 [E]
    const long long* ei   = (const long long*)d_in[1];
    const float*     attr = (const float*)d_in[2];
    float*           out  = (float*)d_out;
    const int E = in_sizes[1] / 2;

    // workspace layout (128 MB total):
    //   [0,  64MB) : Af32 scatter target; later aliased by M2T (f16, 32MB)
    //   [64, 96MB) : Ah   (A as f16, row major)
    //   [96,128MB) : AhT  (A transposed, f16)
    char* ws = (char*)d_ws;
    float*    Af32 = (float*)ws;
    _Float16* M2T  = (_Float16*)ws;
    _Float16* Ah   = (_Float16*)(ws + ((size_t)64 << 20));
    _Float16* AhT  = (_Float16*)(ws + ((size_t)96 << 20));

    // 1) zero A and out (out then carries th0*P0 + th1*A from scatter)
    ppr_zero2<<<(NN * NN / 4) / 256, 256, 0, stream>>>(Af32, out);
    // 2) edge scatter
    ppr_scatter<<<(E + 255) / 256, 256, 0, stream>>>(ei, attr, Af32, out, E);
    // 3) f16 convert + transpose
    ppr_convert<<<dim3(NN / 64, NN / 64), 256, 0, stream>>>(Af32, Ah, AhT);
    // 4) M2 = thr(A@A); out += th2*M2; store M2^T f16 (aliases dead Af32)
    ppr_gemm<true><<<dim3(NN / 128, NN / 128), 256, 0, stream>>>(
        Ah, AhT, M2T, out, TH2);
    // 5) out += th3*thr(A@M2)
    ppr_gemm<false><<<dim3(NN / 128, NN / 128), 256, 0, stream>>>(
        Ah, M2T, nullptr, out, TH3);
}
